// MultiScaleLMN_8933531976481
// MI455X (gfx1250) — matrix operand order repaired and checked by reading, not compiled, not run on hardware
//
#include <hip/hip_runtime.h>
#include <hip/hip_bf16.h>
#include <math.h>

#define T_STEPS 256
#define B_DIM   64
#define IN_DIM  512
#define HID_DIM 1024
#define MEM_DIM 256
#define NMOD_N  8
#define M_DIM   2048

typedef __attribute__((ext_vector_type(16))) __bf16 v16bf;
typedef __attribute__((ext_vector_type(8)))  __bf16 v8bf;
typedef __attribute__((ext_vector_type(8)))  float  v8f;

// ---------------- WMMA fragment helpers (CDNA5 wave32 layouts) ----------------

// A (16x32 bf16): lane L holds row (L&15); K-groups {0..7,16..23} for L<16,
// {8..15,24..31} for L>=16  -> two contiguous 8-element (16B) runs.
static __device__ __forceinline__ v16bf load_a(const __bf16* __restrict__ A,
                                               int lda, int row, int k0, int lane) {
  const int half = lane >> 4;
  const __bf16* base = A + (size_t)row * lda + k0 + half * 8;
  const v8bf b0 = *(const v8bf*)(base);
  const v8bf b1 = *(const v8bf*)(base + 16);
  v16bf a;
#pragma unroll
  for (int i = 0; i < 8; ++i) { a[i] = b0[i]; a[8 + i] = b1[i]; }
  return a;
}

// B (32x16 bf16): lane L holds column n0+(L&15); K-groups {0..7,16..23} for L<16,
// {8..15,24..31} for L>=16 (same K order as A).
static __device__ __forceinline__ v16bf load_b(const __bf16* __restrict__ Bm,
                                               int ldb, int k0, int n0, int lane) {
  const int n = n0 + (lane & 15);
  const int kb = k0 + (lane >> 4) * 8;
  v16bf b;
#pragma unroll
  for (int i = 0; i < 8; ++i) {
    b[i] = Bm[(size_t)(kb + i) * ldb + n];
    b[8 + i] = Bm[(size_t)(kb + 16 + i) * ldb + n];
  }
  return b;
}

static __device__ __forceinline__ v8f wmma_bf16(v16bf a, v16bf b, v8f c) {
  return __builtin_amdgcn_wmma_f32_16x16x32_bf16(false, a, false, b, (short)0, c,
                                                 false, false);
}

// ---------------- per-step kernel 1: h = tanh(x_t*Wxh + m*Wmh^T + bh) ----------------
// grid: HID/128 blocks of 128 threads (4 waves); each wave: 64 rows x 32 cols
// (acc[4][2]) -> every A fragment feeds 2 independent WMMAs (hazard filling).
__global__ void __launch_bounds__(128) h_kernel(const __bf16* __restrict__ xtB,
                                                const __bf16* __restrict__ mB_prev,
                                                const __bf16* __restrict__ WxhB,
                                                const __bf16* __restrict__ WmhT,
                                                const float* __restrict__ bh,
                                                __bf16* __restrict__ hB) {
  const int lane = threadIdx.x & 31;
  const int wave = threadIdx.x >> 5;
  const int n0 = blockIdx.x * 128 + wave * 32;
  const int arow = lane & 15;
  v8f acc[4][2] = {};

  for (int k0 = 0; k0 < IN_DIM; k0 += 32) {              // x_t[64,512] * Wxh[512,1024]
    const v16bf b0 = load_b(WxhB, HID_DIM, k0, n0, lane);
    const v16bf b1 = load_b(WxhB, HID_DIM, k0, n0 + 16, lane);
#pragma unroll
    for (int rt = 0; rt < 4; ++rt) {
      const v16bf a = load_a(xtB, IN_DIM, rt * 16 + arow, k0, lane);
      acc[rt][0] = wmma_bf16(a, b0, acc[rt][0]);
      acc[rt][1] = wmma_bf16(a, b1, acc[rt][1]);
    }
  }
  for (int k0 = 0; k0 < M_DIM; k0 += 32) {               // m[64,2048] * Wmh^T[2048,1024]
    const v16bf b0 = load_b(WmhT, HID_DIM, k0, n0, lane);
    const v16bf b1 = load_b(WmhT, HID_DIM, k0, n0 + 16, lane);
#pragma unroll
    for (int rt = 0; rt < 4; ++rt) {
      const v16bf a = load_a(mB_prev, M_DIM, rt * 16 + arow, k0, lane);
      acc[rt][0] = wmma_bf16(a, b0, acc[rt][0]);
      acc[rt][1] = wmma_bf16(a, b1, acc[rt][1]);
    }
  }

  const int half = lane >> 4;
#pragma unroll
  for (int ct = 0; ct < 2; ++ct) {
    const int col = n0 + ct * 16 + (lane & 15);
    const float bias = bh[col];
#pragma unroll
    for (int rt = 0; rt < 4; ++rt) {
#pragma unroll
      for (int j = 0; j < 8; ++j) {
        const int row = rt * 16 + half * 8 + j;
        hB[(size_t)row * HID_DIM + col] = (__bf16)tanhf(acc[rt][ct][j] + bias);
      }
    }
  }
}

// ---------------- per-step kernel 2: clockwork memory update ----------------
// grid: M/128 blocks of 128 threads. Active blocks: upd = m*Wmm^T + h*Whm^T + bm
// (masked Wmm^T K-loop starts at module*256). Inactive: carry state.
// Writes fp32 master state + bf16 shadow (next step's A operand) + fp32 out slice.
__global__ void __launch_bounds__(128) m_kernel(const float* __restrict__ mF_prev,
                                                const __bf16* __restrict__ mB_prev,
                                                const __bf16* __restrict__ hB,
                                                const __bf16* __restrict__ WmmT,
                                                const __bf16* __restrict__ WhmT,
                                                const float* __restrict__ bm,
                                                float* __restrict__ mF_next,
                                                __bf16* __restrict__ mB_next,
                                                float* __restrict__ out_t,
                                                int m_act) {
  const int c0 = blockIdx.x * 128;

  if (c0 >= m_act) {  // inactive modules: pure state carry (block-uniform branch)
    for (int idx = threadIdx.x; idx < 64 * 32; idx += 128) {
      const int r = idx >> 5;
      const int c = (idx & 31) << 2;
      const size_t o = (size_t)r * M_DIM + c0 + c;
      const float4 v = *(const float4*)(mF_prev + o);
      *(float4*)(mF_next + o) = v;
      *(float4*)(out_t + o) = v;
      mB_next[o + 0] = (__bf16)v.x;
      mB_next[o + 1] = (__bf16)v.y;
      mB_next[o + 2] = (__bf16)v.z;
      mB_next[o + 3] = (__bf16)v.w;
    }
    return;
  }

  const int lane = threadIdx.x & 31;
  const int wave = threadIdx.x >> 5;
  const int n0 = c0 + wave * 32;
  const int arow = lane & 15;
  const int kstart = (c0 / MEM_DIM) * MEM_DIM;  // structural mask: k >= module*256
  v8f acc[4][2] = {};

  for (int k0 = kstart; k0 < M_DIM; k0 += 32) {          // m * Wmm^T (masked)
    const v16bf b0 = load_b(WmmT, M_DIM, k0, n0, lane);
    const v16bf b1 = load_b(WmmT, M_DIM, k0, n0 + 16, lane);
#pragma unroll
    for (int rt = 0; rt < 4; ++rt) {
      const v16bf a = load_a(mB_prev, M_DIM, rt * 16 + arow, k0, lane);
      acc[rt][0] = wmma_bf16(a, b0, acc[rt][0]);
      acc[rt][1] = wmma_bf16(a, b1, acc[rt][1]);
    }
  }
  for (int k0 = 0; k0 < HID_DIM; k0 += 32) {             // h * Whm^T
    const v16bf b0 = load_b(WhmT, M_DIM, k0, n0, lane);
    const v16bf b1 = load_b(WhmT, M_DIM, k0, n0 + 16, lane);
#pragma unroll
    for (int rt = 0; rt < 4; ++rt) {
      const v16bf a = load_a(hB, HID_DIM, rt * 16 + arow, k0, lane);
      acc[rt][0] = wmma_bf16(a, b0, acc[rt][0]);
      acc[rt][1] = wmma_bf16(a, b1, acc[rt][1]);
    }
  }

  const int half = lane >> 4;
#pragma unroll
  for (int ct = 0; ct < 2; ++ct) {
    const int col = n0 + ct * 16 + (lane & 15);
    const float bias = bm[col];
#pragma unroll
    for (int rt = 0; rt < 4; ++rt) {
#pragma unroll
      for (int j = 0; j < 8; ++j) {
        const int row = rt * 16 + half * 8 + j;
        const float v = acc[rt][ct][j] + bias;
        const size_t o = (size_t)row * M_DIM + col;
        mF_next[o] = v;
        mB_next[o] = (__bf16)v;
        out_t[o] = v;
      }
    }
  }
}

// ---------------- one-time prep (fp32 -> bf16, B[K][N] layout) ----------------
__global__ void cvt_kernel(const float* __restrict__ in, __bf16* __restrict__ out, int n) {
  const int i = blockIdx.x * blockDim.x + threadIdx.x;
  if (i < n) out[i] = (__bf16)in[i];
}
// out[k*N+n] = in[n*K+k]  (in is [N,K] row-major)
__global__ void cvt_t_kernel(const float* __restrict__ in, __bf16* __restrict__ out,
                             int K, int N) {
  const int i = blockIdx.x * blockDim.x + threadIdx.x;
  if (i < K * N) {
    const int k = i / N, n = i % N;
    out[i] = (__bf16)in[(size_t)n * K + k];
  }
}
// Wmm^T with block-upper-triangular mask re-applied: nonzero iff k >= (n/256)*256
__global__ void cvt_t_mask_kernel(const float* __restrict__ in, __bf16* __restrict__ out) {
  const int i = blockIdx.x * blockDim.x + threadIdx.x;
  if (i < M_DIM * M_DIM) {
    const int k = i / M_DIM, n = i % M_DIM;
    const float v = (k >= ((n >> 8) << 8)) ? in[(size_t)n * M_DIM + k] : 0.0f;
    out[i] = (__bf16)v;
  }
}
// init both fp32 master state and bf16 shadow from the m_prev input
__global__ void init_m_kernel(const float* __restrict__ in, float* __restrict__ outF,
                              __bf16* __restrict__ outB, int n) {
  const int i = blockIdx.x * blockDim.x + threadIdx.x;
  if (i < n) { const float v = in[i]; outF[i] = v; outB[i] = (__bf16)v; }
}
__global__ void copy_f32_kernel(const float* __restrict__ in, float* __restrict__ out,
                                int n) {
  const int i = blockIdx.x * blockDim.x + threadIdx.x;
  if (i < n) out[i] = in[i];
}

// ---------------- host driver ----------------
extern "C" void kernel_launch(void* const* d_in, const int* in_sizes, int n_in,
                              void* d_out, int out_size, void* d_ws, size_t ws_size,
                              hipStream_t stream) {
  (void)in_sizes; (void)n_in; (void)out_size; (void)ws_size;
  const float* x   = (const float*)d_in[0];  // [T,B,IN]
  const float* m0  = (const float*)d_in[1];  // [B,M]
  const float* Wxh = (const float*)d_in[2];  // [IN,HID]
  const float* Whm = (const float*)d_in[3];  // [M,HID]
  const float* Wmm = (const float*)d_in[4];  // [M,M]
  const float* Wmh = (const float*)d_in[5];  // [HID,M]
  const float* bm  = (const float*)d_in[6];  // [M]
  const float* bh  = (const float*)d_in[7];  // [HID]
  float* out = (float*)d_out;                // [T,B,M] ++ [B,M]

  char* ws = (char*)d_ws;
  const size_t MB = 1u << 20;
  __bf16* WxhB = (__bf16*)(ws);                    // 1 MB   [IN,HID]
  __bf16* WmhT = (__bf16*)(ws + 1 * MB);           // 4 MB   [M,HID]
  __bf16* WhmT = (__bf16*)(ws + 5 * MB);           // 4 MB   [HID,M]
  __bf16* WmmT = (__bf16*)(ws + 9 * MB);           // 8 MB   [M,M]
  __bf16* xB   = (__bf16*)(ws + 17 * MB);          // 16 MB  [T,B,IN]
  float*  mF0  = (float*) (ws + 33 * MB);          // 512 KB
  float*  mF1  = (float*) (ws + 33 * MB + 512 * 1024);
  __bf16* mB0  = (__bf16*)(ws + 34 * MB);          // 256 KB
  __bf16* mB1  = (__bf16*)(ws + 34 * MB + 256 * 1024);
  __bf16* hB   = (__bf16*)(ws + 34 * MB + 512 * 1024);  // 128 KB
  float*  mF[2] = { mF0, mF1 };
  __bf16* mB[2] = { mB0, mB1 };

  // one-time: bf16 weights (L2-resident) + bf16 x + state init
  {
    const int n1 = IN_DIM * HID_DIM;
    cvt_kernel<<<(n1 + 255) / 256, 256, 0, stream>>>(Wxh, WxhB, n1);
    cvt_t_kernel<<<(M_DIM * HID_DIM + 255) / 256, 256, 0, stream>>>(Wmh, WmhT, M_DIM, HID_DIM);
    cvt_t_kernel<<<(HID_DIM * M_DIM + 255) / 256, 256, 0, stream>>>(Whm, WhmT, HID_DIM, M_DIM);
    cvt_t_mask_kernel<<<(M_DIM * M_DIM + 255) / 256, 256, 0, stream>>>(Wmm, WmmT);
    const int nx = T_STEPS * B_DIM * IN_DIM;
    cvt_kernel<<<(nx + 255) / 256, 256, 0, stream>>>(x, xB, nx);
    init_m_kernel<<<(B_DIM * M_DIM + 255) / 256, 256, 0, stream>>>(m0, mF[0], mB[0],
                                                                   B_DIM * M_DIM);
  }

  // sequential clockwork scan: ping-pong state, 2 WMMA kernels per step
  for (int t = 0; t < T_STEPS; ++t) {
    int k = 0;
    while (k < NMOD_N && (t % (1 << k)) == 0) ++k;
    const int m_act = MEM_DIM * k;
    const __bf16* xtB = xB + (size_t)t * B_DIM * IN_DIM;
    const int p = t & 1, q = (t + 1) & 1;
    h_kernel<<<HID_DIM / 128, 128, 0, stream>>>(xtB, mB[p], WxhB, WmhT, bh, hB);
    m_kernel<<<M_DIM / 128, 128, 0, stream>>>(mF[p], mB[p], hB, WmmT, WhmT, bm,
                                              mF[q], mB[q],
                                              out + (size_t)t * B_DIM * M_DIM, m_act);
  }

  // final memory state (after 256 steps it lives in buffer 0)
  copy_f32_kernel<<<(B_DIM * M_DIM + 255) / 256, 256, 0, stream>>>(
      mF[0], out + (size_t)T_STEPS * B_DIM * M_DIM, B_DIM * M_DIM);
}
